// OuterMean_89902255440871
// MI455X (gfx1250) — matrix operand order repaired and checked by reading, not compiled, not run on hardware
//
#include <hip/hip_runtime.h>
#include <hip/hip_bf16.h>

typedef __attribute__((ext_vector_type(16))) __bf16 v16bf;
typedef __attribute__((ext_vector_type(8)))  float  v8f;

#define DIM_MSA 256
#define DIM_HID 32
#define DIM_PAIR 128
#define NM 256
#define NN 384
#define IT 4
#define JT 4

struct Frag { union { unsigned u[8]; v16bf v; }; };

__device__ __forceinline__ unsigned bf16bits(float f) {
  unsigned u = __float_as_uint(f);
  u += 0x7FFFu + ((u >> 16) & 1u);   // round-to-nearest-even
  return u >> 16;
}
__device__ __forceinline__ unsigned pack2(float lo, float hi) {
  return bf16bits(lo) | (bf16bits(hi) << 16);
}

// LDS byte offset of a shared-memory pointer (flat LDS addr low 32 bits).
__device__ __forceinline__ unsigned lds_off(const void* p) {
  return (unsigned)(uintptr_t)p;
}

// CDNA5 async DMA: global -> LDS, 16B per lane, tracked by ASYNCcnt.
__device__ __forceinline__ void async_ld_b128(unsigned lds_addr,
                                              const void* sbase,
                                              unsigned voff) {
  asm volatile("global_load_async_to_lds_b128 %0, %1, %2"
               :: "v"(lds_addr), "v"(voff), "s"(sbase)
               : "memory");
}
__device__ __forceinline__ void wait_async0() {
  asm volatile("s_wait_asynccnt 0" ::: "memory");
}

// ---------------------------------------------------------------------------
// Kernel 1: LayerNorm + left/right projections + mask, packed as bf16x2 over
// m-pairs.  Block = 64 threads (2 waves), one block per (m-pair p, column n).
// Output layout: buf[p][n][h] : uint = (bf16(m=2p) | bf16(m=2p+1)<<16)
// ---------------------------------------------------------------------------
__global__ __launch_bounds__(64) void proj_kernel(
    const float* __restrict__ x,      // [256][384][256]
    const int*   __restrict__ mask,   // [256][384]
    const float* __restrict__ ln_g, const float* __restrict__ ln_b,
    const float* __restrict__ wl, const float* __restrict__ bl,
    const float* __restrict__ wr, const float* __restrict__ br,
    unsigned* __restrict__ leftP,     // [128][384][32]
    unsigned* __restrict__ rightP)    // [128][384][32]
{
  const int n    = blockIdx.x;
  const int p    = blockIdx.y;
  const int tid  = threadIdx.x;
  const int lane = tid & 31;
  const int wv   = tid >> 5;          // 0,1 -> m = 2p + wv

  __shared__ float xn[2][DIM_MSA];

  const int m = 2 * p + wv;
  const float* xr = x + ((size_t)m * NN + n) * DIM_MSA;

  float v[8];
  float s = 0.f, sq = 0.f;
  #pragma unroll
  for (int k = 0; k < 8; ++k) {
    v[k] = xr[lane * 8 + k];
    s  += v[k];
    sq += v[k] * v[k];
  }
  #pragma unroll
  for (int off = 16; off > 0; off >>= 1) {
    s  += __shfl_xor(s,  off, 32);
    sq += __shfl_xor(sq, off, 32);
  }
  const float mu   = s * (1.f / DIM_MSA);
  const float var  = sq * (1.f / DIM_MSA) - mu * mu;
  const float rstd = rsqrtf(var + 1e-5f);
  #pragma unroll
  for (int k = 0; k < 8; ++k) {
    const int kk = lane * 8 + k;
    xn[wv][kk] = (v[k] - mu) * rstd * ln_g[kk] + ln_b[kk];
  }
  __syncthreads();

  // 64 threads -> (side, h); each computes both m rows of the pair
  const int side = tid >> 5;
  const int h    = tid & 31;
  const float* w  = side ? wr : wl;
  const float* bb = side ? br : bl;
  float accA = 0.f, accB = 0.f;
  #pragma unroll 4
  for (int k = 0; k < DIM_MSA; ++k) {
    const float wvv = w[k * DIM_HID + h];
    accA += xn[0][k] * wvv;
    accB += xn[1][k] * wvv;
  }
  const float mA = (float)mask[(2 * p    ) * NN + n];
  const float mB = (float)mask[(2 * p + 1) * NN + n];
  accA = (accA + bb[h]) * mA;
  accB = (accB + bb[h]) * mB;

  unsigned* dst = side ? rightP : leftP;
  dst[((size_t)p * NN + n) * DIM_HID + h] = pack2(accA, accB);
}

// ---------------------------------------------------------------------------
// Kernel 2: valid-pair counts cnt[i][j] = sum_m mask[m][i]*mask[m][j]
// ---------------------------------------------------------------------------
__global__ __launch_bounds__(256) void cnt_kernel(
    const int* __restrict__ mask, float* __restrict__ cntws)
{
  const int idx = blockIdx.x * 256 + threadIdx.x;
  if (idx >= NN * NN) return;
  const int i = idx / NN, j = idx % NN;
  int c = 0;
  for (int m = 0; m < NM; ++m) c += mask[m * NN + i] * mask[m * NN + j];
  cntws[idx] = (float)c;
}

// ---------------------------------------------------------------------------
// Kernel 3: repack wo [1024][128] f32 -> [512][128] uint of bf16x2 over
// adjacent K rows (matches WMMA B-fragment packing).
// ---------------------------------------------------------------------------
__global__ __launch_bounds__(256) void wopack_kernel(
    const float* __restrict__ wo, unsigned* __restrict__ woP)
{
  const int idx = blockIdx.x * 256 + threadIdx.x;
  if (idx >= 512 * DIM_PAIR) return;
  const int q = idx / DIM_PAIR, pc = idx % DIM_PAIR;
  woP[idx] = pack2(wo[(2 * q) * DIM_PAIR + pc], wo[(2 * q + 1) * DIM_PAIR + pc]);
}

// ---------------------------------------------------------------------------
// Kernel 4: fused outer-product-mean.  Block = 256 threads (8 waves) per
// 4x4 (i,j) tile.  Phase A: O[c,d] = sum_m l*r via WMMA bf16 (K=256 in 8
// steps of 32 m), slabs DMA'd global->LDS with async-to-LDS, double-buffered.
// Phase B: out[pair,p] = Oflat @ wo via WMMA.
// ---------------------------------------------------------------------------
__global__ __launch_bounds__(256) void outer_kernel(
    const unsigned* __restrict__ leftP,   // [128][384][32]
    const unsigned* __restrict__ rightP,  // [128][384][32]
    const unsigned* __restrict__ woP,     // [512][128]
    const float*    __restrict__ bo,      // [128]
    const float*    __restrict__ cntws,   // [384][384]
    float*          __restrict__ out)     // [384][384][128]
{
  __shared__ union {
    struct { unsigned L[16 * IT * 32]; unsigned R[16 * JT * 32]; } buf[2]; // 32KB
    unsigned Ob[16][512];                                                  // 32KB
  } sm;

  const int tid  = threadIdx.x;
  const int lane = tid & 31;
  const int wave = tid >> 5;       // 0..7
  const int ln   = lane & 15;
  const int half = lane >> 4;
  const int i0   = blockIdx.x * IT;
  const int j0   = blockIdx.y * JT;

  // warm wo path for phase B (one prefetch per thread covers all 256KB)
  __builtin_prefetch(woP + (size_t)tid * 256, 0, 1);

  // Phase-A assignment: wave -> one i (ii), two j's (jjA, jjA+1)
  const int ii  = wave >> 1;
  const int jjA = (wave & 1) * 2;

  // issue one k-step's slabs (16 m-pairs x 4 cols x 32 h, both sides) as
  // 16B-per-lane async DMAs: 2 chunks per thread per side.
  auto issue_slab = [&](int ks, int bsel) {
    const int pbase = ks * 16;
    const unsigned ldsL = lds_off(sm.buf[bsel].L);
    const unsigned ldsR = lds_off(sm.buf[bsel].R);
    #pragma unroll
    for (int r = 0; r < 2; ++r) {
      const int c16 = r * 256 + tid;   // 0..511 : 16-byte chunk id
      const int pp  = c16 >> 5;
      const int rem = c16 & 31;
      const int ci  = rem >> 3;
      const int q   = rem & 7;
      const unsigned so = (unsigned)(pp * (IT * 32) + ci * 32 + q * 4) * 4u;
      const unsigned gL = (unsigned)(((pbase + pp) * (NN * DIM_HID) +
                                      (i0 + ci) * DIM_HID + q * 4) * 4);
      const unsigned gR = (unsigned)(((pbase + pp) * (NN * DIM_HID) +
                                      (j0 + ci) * DIM_HID + q * 4) * 4);
      async_ld_b128(ldsL + so, leftP,  gL);
      async_ld_b128(ldsR + so, rightP, gR);
    }
  };

  v8f acc[2][2][2];                 // [jsel][ctile][dtile]
  #pragma unroll
  for (int a = 0; a < 2; ++a)
    #pragma unroll
    for (int b = 0; b < 2; ++b)
      #pragma unroll
      for (int c = 0; c < 2; ++c) acc[a][b][c] = (v8f){};

  issue_slab(0, 0);
  wait_async0();
  __syncthreads();

  for (int ks = 0; ks < NM / 32; ++ks) {
    const int cur = ks & 1;
    if (ks < NM / 32 - 1) issue_slab(ks + 1, cur ^ 1);   // DMA next buffer

    const unsigned* sl = sm.buf[cur].L;
    const unsigned* sr = sm.buf[cur].R;

    Frag a[2], b[2][2];
    #pragma unroll
    for (int ct = 0; ct < 2; ++ct) {
      const int row = ct * 16 + ln;           // c
      #pragma unroll
      for (int v = 0; v < 4; ++v) {
        a[ct].u[v]     = sl[(4 * half + v)     * (IT * 32) + ii * 32 + row];
        a[ct].u[v + 4] = sl[(8 + 4 * half + v) * (IT * 32) + ii * 32 + row];
      }
    }
    #pragma unroll
    for (int js = 0; js < 2; ++js) {
      const int jj = jjA + js;
      #pragma unroll
      for (int dt = 0; dt < 2; ++dt) {
        const int col = dt * 16 + ln;         // d
        #pragma unroll
        for (int v = 0; v < 8; ++v)
          b[js][dt].u[v] = sr[((v >> 2) * 8 + 4 * half + (v & 3)) * (JT * 32) + jj * 32 + col];
      }
    }
    #pragma unroll
    for (int js = 0; js < 2; ++js)
      #pragma unroll
      for (int ct = 0; ct < 2; ++ct)
        #pragma unroll
        for (int dt = 0; dt < 2; ++dt)
          acc[js][ct][dt] = __builtin_amdgcn_wmma_f32_16x16x32_bf16(
              false, a[ct].v, false, b[js][dt].v,
              (short)0, acc[js][ct][dt], false, false);

    wait_async0();      // next buffer's DMA landed
    __syncthreads();    // everyone done reading cur, writing nxt
  }

  // pack O -> LDS bf16x2 over adjacent d (cross-lane pair via shfl_xor 1)
  #pragma unroll
  for (int js = 0; js < 2; ++js) {
    const int pair = ii * 4 + jjA + js;
    #pragma unroll
    for (int ct = 0; ct < 2; ++ct)
      #pragma unroll
      for (int dt = 0; dt < 2; ++dt)
        #pragma unroll
        for (int r = 0; r < 8; ++r) {
          const float mine  = acc[js][ct][dt][r];
          const float other = __shfl_xor(mine, 1, 32);
          if ((lane & 1) == 0) {
            const int c = ct * 16 + r + 8 * half;
            const int d = dt * 16 + ln;       // even
            sm.Ob[pair][(c * 32 + d) >> 1] = pack2(mine, other);
          }
        }
  }
  __syncthreads();

  // Phase B: OUT[16 pairs, 128] = Oflat[16,1024] @ wo[1024,128]
  const int p0 = wave * 16;                   // this wave's N tile
  v8f oacc = (v8f){};
  for (int ks2 = 0; ks2 < 32; ++ks2) {
    Frag a2, b2;
    #pragma unroll
    for (int v = 0; v < 4; ++v) {
      a2.u[v]     = sm.Ob[ln][ks2 * 16 + 4 * half + v];
      a2.u[v + 4] = sm.Ob[ln][ks2 * 16 + 8 + 4 * half + v];
    }
    #pragma unroll
    for (int v = 0; v < 8; ++v)
      b2.u[v] = woP[(ks2 * 16 + (v >> 2) * 8 + 4 * half + (v & 3)) * DIM_PAIR + p0 + ln];
    oacc = __builtin_amdgcn_wmma_f32_16x16x32_bf16(
        false, a2.v, false, b2.v, (short)0, oacc, false, false);
  }

  // epilogue: + bo, / (cnt + eps)
  const float bop = bo[p0 + ln];
  #pragma unroll
  for (int r = 0; r < 8; ++r) {
    const int pair = r + 8 * half;
    const int gi = i0 + (pair >> 2);
    const int gj = j0 + (pair & 3);
    const float cnt = cntws[gi * NN + gj];
    out[((size_t)gi * NN + gj) * DIM_PAIR + p0 + ln] =
        (oacc[r] + bop) / (cnt + 0.001f);
  }
}

// ---------------------------------------------------------------------------
extern "C" void kernel_launch(void* const* d_in, const int* in_sizes, int n_in,
                              void* d_out, int out_size, void* d_ws, size_t ws_size,
                              hipStream_t stream) {
  const float* x    = (const float*)d_in[0];
  const int*   mask = (const int*)  d_in[1];
  const float* ln_g = (const float*)d_in[2];
  const float* ln_b = (const float*)d_in[3];
  const float* wl   = (const float*)d_in[4];
  const float* bl   = (const float*)d_in[5];
  const float* wr   = (const float*)d_in[6];
  const float* br   = (const float*)d_in[7];
  const float* wo   = (const float*)d_in[8];
  const float* bo   = (const float*)d_in[9];
  float* out = (float*)d_out;

  // workspace carve-up (~12.8 MB total)
  unsigned* leftP  = (unsigned*)d_ws;
  unsigned* rightP = leftP  + (size_t)128 * NN * DIM_HID;
  unsigned* woP    = rightP + (size_t)128 * NN * DIM_HID;
  float*    cntws  = (float*)(woP + 512 * DIM_PAIR);

  dim3 g1(NN, NM / 2);
  proj_kernel<<<g1, 64, 0, stream>>>(x, mask, ln_g, ln_b, wl, bl, wr, br,
                                     leftP, rightP);
  cnt_kernel<<<(NN * NN + 255) / 256, 256, 0, stream>>>(mask, cntws);
  wopack_kernel<<<(512 * DIM_PAIR + 255) / 256, 256, 0, stream>>>(wo, woP);

  dim3 g2(NN / IT, NN / JT);
  outer_kernel<<<g2, 256, 0, stream>>>(leftP, rightP, woP, bo, cntws, out);
}
